// Model_87686052315546
// MI455X (gfx1250) — matrix operand order repaired and checked by reading, not compiled, not run on hardware
//
#include <hip/hip_runtime.h>
#include <hip/hip_bf16.h>

typedef _Float16 v16h __attribute__((ext_vector_type(16)));
typedef _Float16 h8   __attribute__((ext_vector_type(8)));
typedef float    v8f  __attribute__((ext_vector_type(8)));

#define NT      11      // tiles per dim
#define TS      256     // tile size
#define STEP    96
#define MD      128     // max disp
#define DD      257     // 2*MD+1 displacements per dim
#define NB      17      // 16x16 output blocks per dim (17*16=272 >= 257)
#define IMG     1216
#define COLS    544     // staged search-window columns (256 + 287 + pad)
#define SSTRIDE 552     // LDS row stride in halves: 16B aligned, bank-rotating
#define RING    32      // ring-buffer rows
#define CHUNK   8       // tile rows (a) per staging chunk
#define TCH     9       // Toeplitz K-chunks per row (covers j in [0,256) for all q)
#define NTHR    544     // 17 waves of 32
#define NEGINF  (-3.0e38f)

// ---------------- Kernel 1: per-tile mean removal + f32->f16 ----------------
__global__ __launch_bounds__(256) void prep_tiles(const float* __restrict__ src,
                                                  unsigned short* __restrict__ TwsRaw) {
    _Float16* Tws = (_Float16*)TwsRaw;
    const int tile = blockIdx.x;
    const int ty = tile / NT, tx = tile % NT;
    const int oy = ty * STEP, ox = tx * STEP;
    __shared__ float red[256];
    const int i = threadIdx.x;                      // tile row
    const float* rp = src + (size_t)(oy + i) * IMG + ox;
    float s = 0.f;
    for (int j = 0; j < TS; ++j) s += rp[j];
    red[i] = s;
    __syncthreads();
    for (int off = 128; off > 0; off >>= 1) {
        if (i < off) red[i] += red[i + off];
        __syncthreads();
    }
    const float mean = red[0] * (1.0f / 65536.0f);
    _Float16* op = Tws + (size_t)tile * (TS * TS) + (size_t)i * TS;
    for (int j = 0; j < TS; ++j) op[j] = (_Float16)(rp[j] - mean);
}

// ------------- Kernel 2: WMMA correlation + per-block argmax ----------------
__global__ __launch_bounds__(NTHR) void corr_wmma(const float* __restrict__ tgt,
                                                  const unsigned short* __restrict__ TwsRaw,
                                                  float* __restrict__ bestVal,
                                                  int* __restrict__ bestIdx) {
    const _Float16* Tws = (const _Float16*)TwsRaw;
    __shared__ __attribute__((aligned(16))) _Float16 ldsS[RING * SSTRIDE];     // 35.3 KB
    __shared__ __attribute__((aligned(16))) _Float16 ldsB[CHUNK * TCH * 512];  // 73.7 KB
    __shared__ float redV[NTHR];
    __shared__ int   redI[NTHR];

    const int by   = blockIdx.x;        // dy block 0..16
    const int tile = blockIdx.y;        // 0..120
    const int ty = tile / NT, tx = tile % NT;
    const int oy = ty * STEP, ox = tx * STEP;
    const int tid  = threadIdx.x;
    const int lane = tid & 31;
    const int wv   = tid >> 5;          // wave id == dx block 0..16
    const int r0   = 16 * by;           // first padded-window row of this stripe
    const _Float16* Tt = Tws + (size_t)tile * (TS * TS);

    // Stage search-window rows [rBase, rBase+nRows) into the ring (pad + f16 cvt).
    auto stageRows = [&](int rBase, int nRows) {
        const int c = tid;              // 0..543, one column per thread
        const int x = ox + c - MD;
        const bool xin = (unsigned)x < (unsigned)IMG;
        for (int rr = 0; rr < nRows; ++rr) {
            const int r = rBase + rr;
            const int y = oy + r - MD;
            float v = 0.f;
            if (xin && (unsigned)y < (unsigned)IMG) v = tgt[(size_t)y * IMG + x];
            ldsS[(r & (RING - 1)) * SSTRIDE + c] = (_Float16)v;
        }
    };
    // Build pre-swizzled Toeplitz B fragments for rows a = abase..abase+7.
    // B[k][q] = Trow[32t + k - q]; per-lane layout: n=lane&15, K = (lane<16?0:16)+e.
    auto buildB = [&](int abase) {
        for (int idx = tid; idx < CHUNK * TCH * 512; idx += NTHR) {
            const int f  = idx & 511;
            const int tq = idx >> 9;
            const int ia = tq / TCH;
            const int t  = tq - ia * TCH;
            const int ln = f >> 4, e = f & 15;
            const int q  = ln & 15;
            const int kb = (ln < 16) ? 0 : 16;
            const int j  = 32 * t + kb + e - q;
            _Float16 v = (_Float16)0.f;
            if (j >= 0 && j < TS) v = Tt[(size_t)(abase + ia) * TS + j];
            ldsB[idx] = v;
        }
    };

    stageRows(r0, RING);
    buildB(0);
    __syncthreads();

    v8f acc = {0.f, 0.f, 0.f, 0.f, 0.f, 0.f, 0.f, 0.f};
    const int m       = lane & 15;                 // A-matrix M (dy offset)
    const int base0   = (lane < 16) ? 0 : 16;      // A-matrix K base (contiguous, matches B)
    const int colbase = 16 * wv + base0;           // dx0 + K base

    for (int abase = 0; abase < TS; abase += CHUNK) {
        for (int ia = 0; ia < CHUNK; ++ia) {
            const int a = abase + ia;
            const _Float16* rowp = &ldsS[((r0 + a + m) & (RING - 1)) * SSTRIDE + colbase];
            const _Float16* bp   = &ldsB[ia * TCH * 512 + lane * 16];
#pragma unroll
            for (int t = 0; t < TCH; ++t) {
                h8 a0 = *(const h8*)(rowp + 32 * t);        // K = base0+0..7
                h8 a1 = *(const h8*)(rowp + 32 * t + 8);    // K = base0+8..15
                h8 b0 = *(const h8*)(bp + 512 * t);
                h8 b1 = *(const h8*)(bp + 512 * t + 8);
                v16h av = __builtin_shufflevector(a0, a1, 0,1,2,3,4,5,6,7,8,9,10,11,12,13,14,15);
                v16h bv = __builtin_shufflevector(b0, b1, 0,1,2,3,4,5,6,7,8,9,10,11,12,13,14,15);
                acc = __builtin_amdgcn_wmma_f32_16x16x32_f16(
                        false, av, false, bv, (short)0, acc, false, false);
            }
        }
        __syncthreads();                            // all reads done before restage
        if (abase + CHUNK < TS) {
            stageRows(r0 + abase + RING, CHUNK);    // rows one chunk ahead
            buildB(abase + CHUNK);
            __syncthreads();
        }
    }

    // Per-lane argmax over the 8 accumulator rows (deterministic tie-break: min idx).
    float bv_ = NEGINF; int bi_ = 0x7fffffff;
    const int dx = 16 * wv + (lane & 15);
#pragma unroll
    for (int g = 0; g < 8; ++g) {
        const int dy = 16 * by + g + ((lane < 16) ? 0 : 8);   // D-matrix M mapping
        const bool valid = (dy < DD) && (dx < DD);
        const float v = valid ? acc[g] : NEGINF;
        const int idx = dy * DD + dx;
        if (v > bv_ || (v == bv_ && idx < bi_)) { bv_ = v; bi_ = idx; }
    }
    redV[tid] = bv_; redI[tid] = bi_;
    __syncthreads();
    if (lane == 0) {
        float B = redV[tid]; int I = redI[tid];
        for (int l = 1; l < 32; ++l) {
            const float v = redV[tid + l]; const int i2 = redI[tid + l];
            if (v > B || (v == B && i2 < I)) { B = v; I = i2; }
        }
        const int o = (tile * NB + by) * NB + wv;
        bestVal[o] = B; bestIdx[o] = I;
    }
}

// ---------------- Kernel 3: per-tile final reduce + scale -------------------
__global__ void final_reduce(const float* __restrict__ bestVal,
                             const int* __restrict__ bestIdx,
                             float* __restrict__ out) {
    const int tile = blockIdx.x * blockDim.x + threadIdx.x;
    if (tile >= NT * NT) return;
    const float* bv = bestVal + tile * (NB * NB);
    const int*   bi = bestIdx + tile * (NB * NB);
    float B = NEGINF; int I = 0x7fffffff;
    for (int k = 0; k < NB * NB; ++k) {
        const float v = bv[k]; const int i2 = bi[k];
        if (v > B || (v == B && i2 < I)) { B = v; I = i2; }
    }
    const float dy = (float)(I / DD - MD);
    const float dx = (float)(I % DD - MD);
    const float sc = 2.0f / (float)IMG;
    out[tile * 2 + 0] = dx * sc;   // channel order (dx, dy), layout (1,11,11,2)
    out[tile * 2 + 1] = dy * sc;
}

extern "C" void kernel_launch(void* const* d_in, const int* in_sizes, int n_in,
                              void* d_out, int out_size, void* d_ws, size_t ws_size,
                              hipStream_t stream) {
    const float* src = (const float*)d_in[0];
    const float* tgt = (const float*)d_in[1];
    float* out = (float*)d_out;

    // Workspace layout (assumes ws_size >= ~16.2 MB):
    //  [0, 15,859,712)             : 121 zero-mean f16 tiles (121*256*256*2 B)
    //  [TWS, TWS + 139,876)        : per-block best value  (121*17*17 f32)
    //  [.., + 139,876)             : per-block best index  (121*17*17 i32)
    const size_t TWS = (size_t)NT * NT * TS * TS * sizeof(unsigned short);
    unsigned short* Tws = (unsigned short*)d_ws;
    float* bestVal = (float*)((char*)d_ws + TWS);
    int*   bestIdx = (int*)((char*)d_ws + TWS + (size_t)NT * NT * NB * NB * sizeof(float));

    prep_tiles<<<NT * NT, 256, 0, stream>>>(src, Tws);
    corr_wmma<<<dim3(NB, NT * NT), NTHR, 0, stream>>>(tgt, Tws, bestVal, bestIdx);
    final_reduce<<<2, 64, 0, stream>>>(bestVal, bestIdx, out);
}
